// ItemConv_55697135895081
// MI455X (gfx1250) — matrix operand order repaired and checked by reading, not compiled, not run on hardware
//
#include <hip/hip_runtime.h>
#include <hip/hip_bf16.h>

// ---------- vector types matching the gfx1250 WMMA builtin signatures ----------
typedef __bf16 v16bf __attribute__((ext_vector_type(16)));
typedef __bf16 v8bf  __attribute__((ext_vector_type(8)));
typedef float  v8f   __attribute__((ext_vector_type(8)));

// Split a float8 into bf16 hi + bf16 lo (x ~= hi + lo), for 3xBF16 fp32-emulated WMMA.
__device__ __forceinline__ void cvt_split(const v8f& a, v8bf& hi, v8bf& lo) {
    hi = __builtin_convertvector(a, v8bf);
    v8f hf = __builtin_convertvector(hi, v8f);
    lo = __builtin_convertvector(a - hf, v8bf);
}

__device__ __forceinline__ v16bf combine8(v8bf a, v8bf b) {
    return __builtin_shufflevector(a, b, 0,1,2,3,4,5,6,7,8,9,10,11,12,13,14,15);
}

// =====================================================================
// GEMM: Y[nrows,NCOLS] = X[nrows,KDIM] @ W[KDIM,NCOLS]   (fp32 in/out)
// MODE 0: plain.  MODE 1: Y = relu(X@W + X)  (requires KDIM==NCOLS)
// 256 threads = 8 waves; each wave owns a 16-row strip x all NCOLS cols.
// 3x bf16 WMMA per (ktile, ntile) recovers ~fp32 accuracy.
// =====================================================================
template<int KDIM, int NCOLS, int MODE>
__global__ __launch_bounds__(256)
void gemm_bf16x3(const float* __restrict__ X, const float* __restrict__ W,
                 float* __restrict__ Y, int nrows)
{
    // W staged transposed + pre-split: WT[col][k], so a B fragment
    // (16 contiguous K per lane-half) is one 32B LDS load.
    __shared__ __bf16 WhiT[NCOLS * KDIM];
    __shared__ __bf16 WloT[NCOLS * KDIM];

    const int tid = threadIdx.x;
    for (int idx = tid; idx < KDIM * NCOLS; idx += 256) {
        const int k = idx / NCOLS, c = idx % NCOLS;
        const float w = W[idx];
        const __bf16 h = (__bf16)w;
        WhiT[c * KDIM + k] = h;
        WloT[c * KDIM + k] = (__bf16)(w - (float)h);
    }
    __syncthreads();

    const int wave = tid >> 5, lane = tid & 31;
    const int half = lane >> 4, r = lane & 15;
    const int rowBase = (blockIdx.x * 8 + wave) * 16;
    int arow = rowBase + r;                 // A-matrix row for this lane
    if (arow >= nrows) arow = nrows - 1;    // clamp: EXEC stays all-1s for WMMA
    const float* xrow = X + (size_t)arow * KDIM;

    v8f acc[NCOLS / 16] = {};

    #pragma unroll
    for (int kk = 0; kk < KDIM / 32; ++kk) {
        const int kb = kk * 32;
        // A fragment (16x32 bf16): elems 0..15 = K kb+half*16 .. +15, contiguous
        const v8f a0 = *(const v8f*)(xrow + kb + half * 16);
        const v8f a1 = *(const v8f*)(xrow + kb + half * 16 + 8);
        v8bf a0h, a0l, a1h, a1l;
        cvt_split(a0, a0h, a0l);
        cvt_split(a1, a1h, a1l);
        const v16bf ahi = combine8(a0h, a1h);
        const v16bf alo = combine8(a0l, a1l);

        #pragma unroll
        for (int nt = 0; nt < NCOLS / 16; ++nt) {
            const int c = nt * 16 + r;
            // B fragment (32x16 bf16): lane-half h holds K = kb + h*16 .. +15, contiguous.
            const int base = c * KDIM + kb + half * 16;
            const v16bf bhi = *(const v16bf*)(&WhiT[base]);
            const v16bf blo = *(const v16bf*)(&WloT[base]);
            acc[nt] = __builtin_amdgcn_wmma_f32_16x16x32_bf16(
                false, ahi, false, bhi, (short)0, acc[nt], false, false);
            acc[nt] = __builtin_amdgcn_wmma_f32_16x16x32_bf16(
                false, ahi, false, blo, (short)0, acc[nt], false, false);
            acc[nt] = __builtin_amdgcn_wmma_f32_16x16x32_bf16(
                false, alo, false, bhi, (short)0, acc[nt], false, false);
        }
    }

    // C/D layout: VGPR j -> row rowBase + half*8 + j, col nt*16 + r
    #pragma unroll
    for (int nt = 0; nt < NCOLS / 16; ++nt) {
        const int c = nt * 16 + r;
        #pragma unroll
        for (int j = 0; j < 8; ++j) {
            const int orow = rowBase + half * 8 + j;
            if (orow < nrows) {
                float v = acc[nt][j];
                if (MODE == 1) {
                    v += X[(size_t)orow * KDIM + c];
                    v = v > 0.f ? v : 0.f;
                }
                Y[(size_t)orow * NCOLS + c] = v;
            }
        }
    }
}

// =====================================================================
// COO SpMM scatter: Z[row] += val * A[col]   (one wave per edge, float4/lane)
// =====================================================================
__global__ __launch_bounds__(256)
void spmm_scatter(const int* __restrict__ erow, const int* __restrict__ ecol,
                  const float* __restrict__ eval, const float* __restrict__ A,
                  float* __restrict__ Z, int nedges)
{
    const int lane = threadIdx.x & 31;
    const int e = (int)((blockIdx.x * 256 + threadIdx.x) >> 5);
    if (e >= nedges) return;
    const int r = erow[e], c = ecol[e];
    const float v = eval[e];
    const float4 x = *(const float4*)(A + (size_t)c * 128 + lane * 4);
    float* zp = Z + (size_t)r * 128 + lane * 4;
    unsafeAtomicAdd(zp + 0, v * x.x);
    unsafeAtomicAdd(zp + 1, v * x.y);
    unsafeAtomicAdd(zp + 2, v * x.z);
    unsafeAtomicAdd(zp + 3, v * x.w);
}

// =====================================================================
// Row softmax over K=64 (in place) + pooled column-normalization weights:
//   H1w[n,k] = H1s[n,k] * adj[n] / (adj[n]*sum_k H1s[n,k] + 1e-8)
// One wave per row; 2 elements per lane; shuffle reductions.
// =====================================================================
__global__ __launch_bounds__(256)
void softmax_pool(float* __restrict__ S, const float* __restrict__ adj,
                  float* __restrict__ H1w, int nrows)
{
    const int wave = threadIdx.x >> 5, lane = threadIdx.x & 31;
    const int n = blockIdx.x * 8 + wave;
    if (n >= nrows) return;
    float* srow = S + (size_t)n * 64;
    float e0 = srow[lane], e1 = srow[lane + 32];
    float m = fmaxf(e0, e1);
    #pragma unroll
    for (int o = 16; o; o >>= 1) m = fmaxf(m, __shfl_xor(m, o, 32));
    e0 = __expf(e0 - m); e1 = __expf(e1 - m);
    float s = e0 + e1;
    #pragma unroll
    for (int o = 16; o; o >>= 1) s += __shfl_xor(s, o, 32);
    const float inv = 1.f / s;
    const float p0 = e0 * inv, p1 = e1 * inv;
    srow[lane] = p0; srow[lane + 32] = p1;
    float sp = p0 + p1;
    #pragma unroll
    for (int o = 16; o; o >>= 1) sp += __shfl_xor(sp, o, 32);
    const float a = adj[n];
    const float w = a / (sp * a + 1e-8f);
    H1w[(size_t)n * 64 + lane]      = p0 * w;
    H1w[(size_t)n * 64 + lane + 32] = p1 * w;
}

// =====================================================================
// hx[64,128] = sum_n H1w[n,:]^T (outer) Xc[n,:]
// Block covers 128 nodes in 4 LDS-staged chunks of 32; thread (k = tid&63,
// dbase = (tid>>6)*32) owns 32 accumulators; 8192 global atomics per block.
// =====================================================================
__global__ __launch_bounds__(256)
void pool_reduce(const float* __restrict__ H1w, const float* __restrict__ Xc,
                 float* __restrict__ hx, int nrows)
{
    __shared__ float sh_h[32 * 64];
    __shared__ float sh_x[32 * 128];
    const int tid = threadIdx.x;
    const int k = tid & 63;
    const int dbase = (tid >> 6) * 32;
    float acc[32] = {};
    const int node0 = blockIdx.x * 128;

    for (int ch = 0; ch < 4; ++ch) {
        const int nb = node0 + ch * 32;
        for (int idx = tid; idx < 32 * 64; idx += 256) {
            const int nn = nb + (idx >> 6);
            sh_h[idx] = (nn < nrows) ? H1w[(size_t)nn * 64 + (idx & 63)] : 0.f;
        }
        for (int idx = tid; idx < 32 * 128; idx += 256) {
            const int nn = nb + (idx >> 7);
            sh_x[idx] = (nn < nrows) ? Xc[(size_t)nn * 128 + (idx & 127)] : 0.f;
        }
        __syncthreads();
        #pragma unroll 4
        for (int i = 0; i < 32; ++i) {
            const float a = sh_h[i * 64 + k];
            const float* xr = &sh_x[i * 128 + dbase];
            #pragma unroll
            for (int j = 0; j < 32; ++j) acc[j] = fmaf(a, xr[j], acc[j]);
        }
        __syncthreads();
    }
    float* out = hx + k * 128 + dbase;
    #pragma unroll
    for (int j = 0; j < 32; ++j) unsafeAtomicAdd(out + j, acc[j]);
}

// =====================================================================
// Finalize layer: xn = G + Z; X <- xn; out_item += l2n(xn)/3; out_hs += l2n(G)/2
// One wave per row, float4 per lane, shuffle reductions for the norms.
// =====================================================================
__global__ __launch_bounds__(256)
void finalize_layer(const float* __restrict__ G, const float* __restrict__ Z,
                    float* __restrict__ Xout, float* __restrict__ out_item,
                    float* __restrict__ out_hs, int nrows)
{
    const int wave = threadIdx.x >> 5, lane = threadIdx.x & 31;
    const int n = blockIdx.x * 8 + wave;
    if (n >= nrows) return;
    const size_t off = (size_t)n * 128 + lane * 4;
    const float4 g = *(const float4*)(G + off);
    const float4 z = *(const float4*)(Z + off);
    float4 x;
    x.x = g.x + z.x; x.y = g.y + z.y; x.z = g.z + z.z; x.w = g.w + z.w;
    *(float4*)(Xout + off) = x;

    float ng = g.x*g.x + g.y*g.y + g.z*g.z + g.w*g.w;
    float nx = x.x*x.x + x.y*x.y + x.z*x.z + x.w*x.w;
    #pragma unroll
    for (int o = 16; o; o >>= 1) {
        ng += __shfl_xor(ng, o, 32);
        nx += __shfl_xor(nx, o, 32);
    }
    const float ig = 0.5f      / fmaxf(sqrtf(ng), 1e-12f);   // /LAYERS
    const float ix = (1.f/3.f) / fmaxf(sqrtf(nx), 1e-12f);   // /(LAYERS+1)

    float4 oi = *(const float4*)(out_item + off);
    oi.x += x.x * ix; oi.y += x.y * ix; oi.z += x.z * ix; oi.w += x.w * ix;
    *(float4*)(out_item + off) = oi;
    float4 oh = *(const float4*)(out_hs + off);
    oh.x += g.x * ig; oh.y += g.y * ig; oh.z += g.z * ig; oh.w += g.w * ig;
    *(float4*)(out_hs + off) = oh;
}

__global__ __launch_bounds__(256)
void init_kernel(const float* __restrict__ emb, float* __restrict__ X,
                 float* __restrict__ out_item, float* __restrict__ out_hs, int total)
{
    const int i = blockIdx.x * 256 + threadIdx.x;
    if (i < total) {
        const float e = emb[i];
        X[i] = e;
        out_item[i] = e * (1.f / 3.f);   // final[0] = raw embedding
        out_hs[i] = 0.f;
    }
}

__global__ __launch_bounds__(256)
void zero_kernel(float* __restrict__ p, int total)
{
    const int i = blockIdx.x * 256 + threadIdx.x;
    if (i < total) p[i] = 0.f;
}

// =====================================================================
extern "C" void kernel_launch(void* const* d_in, const int* in_sizes, int n_in,
                              void* d_out, int out_size, void* d_ws, size_t ws_size,
                              hipStream_t stream)
{
    const float* adj  = (const float*)d_in[0];
    const int*   erow = (const int*)  d_in[1];
    const int*   ecol = (const int*)  d_in[2];
    const float* eval = (const float*)d_in[3];
    const float* emb  = (const float*)d_in[4];
    const float* Wit[2] = { (const float*)d_in[5], (const float*)d_in[6] };
    const float* W_i1 = (const float*)d_in[7];
    const float* W_i2 = (const float*)d_in[8];
    // d_in[9] = channel (always 0 path in reference)

    const int N = in_sizes[0];
    const int E = in_sizes[1];
    const int D = 128, K = 64;
    const int total = N * D;

    // Workspace layout (floats): X | A | Z | hx. The [N,64] softmax buffers
    // P (=H1s) and Q (=H1w) live inside X, which is dead between the first
    // GEMM of a layer and the finalize that rewrites it.
    float* ws = (float*)d_ws;
    float* X  = ws;
    float* A  = ws + (size_t)total;
    float* Z  = ws + (size_t)2 * total;
    float* hx = ws + (size_t)3 * total;
    float* P  = X;                       // [N,64] softmax probs
    float* Q  = X + (size_t)N * K;       // [N,64] normalized pooling weights

    float* out_item = (float*)d_out;
    float* out_hs   = (float*)d_out + (size_t)total;

    const int eltGrid  = (total + 255) / 256;
    const int gemmGrid = (N + 127) / 128;
    const int rowGrid  = (N + 7) / 8;

    init_kernel<<<eltGrid, 256, 0, stream>>>(emb, X, out_item, out_hs, total);

    for (int layer = 0; layer < 2; ++layer) {
        // A = x @ W_item
        gemm_bf16x3<128, 128, 0><<<gemmGrid, 256, 0, stream>>>(X, Wit[layer], A, N);
        // Z = spmm(A)
        zero_kernel<<<eltGrid, 256, 0, stream>>>(Z, total);
        spmm_scatter<<<(E + 7) / 8, 256, 0, stream>>>(erow, ecol, eval, A, Z, E);
        // A = relu(Z @ W_i1 + Z)           [H1]
        gemm_bf16x3<128, 128, 1><<<gemmGrid, 256, 0, stream>>>(Z, W_i1, A, N);
        // P = H1 @ W_i2                    [N,64]
        gemm_bf16x3<128, 64, 0><<<gemmGrid, 256, 0, stream>>>(A, W_i2, P, N);
        // P = softmax(P); Q = P * adj/(adj*sum+1e-8)
        softmax_pool<<<rowGrid, 256, 0, stream>>>(P, adj, Q, N);
        // hx = Q^T @ Z                     [64,128]
        zero_kernel<<<(K * D + 255) / 256, 256, 0, stream>>>(hx, K * D);
        pool_reduce<<<gemmGrid, 256, 0, stream>>>(Q, Z, hx, N);
        // A = P @ hx                       [h in the reference]
        gemm_bf16x3<64, 128, 0><<<gemmGrid, 256, 0, stream>>>(P, hx, A, N);
        // X = A + Z; accumulate normalized outputs
        finalize_layer<<<rowGrid, 256, 0, stream>>>(A, Z, X, out_item, out_hs, N);
    }
}
